// RNN_language_model_28750511079515
// MI455X (gfx1250) — matrix operand order repaired and checked by reading, not compiled, not run on hardware
//
#include <hip/hip_runtime.h>
#include <hip/hip_bf16.h>
#include <math.h>

// ---- problem constants ----
#define TT   128     // timesteps (S-1)
#define BB   32      // batch
#define HH   1024    // hidden
#define VV   32000   // vocab
#define SS   129     // seq len incl target shift
#define G4   4096    // 4*H (gate width)
#define MR   4096    // T*B rows
#define KC   64      // K chunk staged per LDS stage

typedef __bf16 bf16_t;
typedef __attribute__((ext_vector_type(16))) __bf16 v16bf;
typedef __attribute__((ext_vector_type(8)))  __bf16 v8bf;
typedef __attribute__((ext_vector_type(8)))  float  v8f;
typedef __attribute__((ext_vector_type(4)))  unsigned int v4u;
typedef __attribute__((ext_vector_type(8)))  int v8i;
typedef __attribute__((ext_vector_type(4)))  int v4i;

union ABFrag { v16bf v; v8bf h[2]; };

__device__ __forceinline__ float sigf(float x) { return 1.0f / (1.0f + __expf(-x)); }

// ---------------------------------------------------------------------------
// TDM: async 2D tile DMA global -> LDS (bf16 elements), with LDS row padding
// of 4 DWORDs (8 bf16) every 32 DWORDs (64 bf16) to match the +8 padded LDS
// rows used by the WMMA fragment loads. Tracked by TENSORcnt.
// D# layout per CDNA5 ISA. 6-arg builtin (clang-23).
// ---------------------------------------------------------------------------
__device__ __forceinline__ void tdm_load_tile_bf16(unsigned lds_off, const void* gaddr,
                                                   unsigned tensor_d0, unsigned tensor_d1,
                                                   unsigned tile_d0, unsigned tile_d1,
                                                   unsigned long long stride0_elems) {
    unsigned long long ga = (unsigned long long)(uintptr_t)gaddr;
    v4u g0;
    g0[0] = 1u;                                            // count=1 valid descriptor
    g0[1] = lds_off;                                       // LDS byte address
    g0[2] = (unsigned)(ga & 0xffffffffu);                  // global_addr[31:0]
    g0[3] = (unsigned)((ga >> 32) & 0x01ffffffu) | (2u << 30); // addr[56:32] | type=2
    v8i g1;
    g1[0] = (int)((1u << 16)      // data_size = 1 -> 2 bytes (bf16)
                | (1u << 20)      // pad_enable
                | (4u << 22)      // pad_interval: 32 DWORDs (= 64 bf16 row)
                | (3u << 25));    // pad_amount:   4 DWORDs (=  8 bf16 pad)
    g1[1] = (int)((tensor_d0 & 0xffffu) << 16);            // dim0[15:0]
    g1[2] = (int)(((tensor_d0 >> 16) & 0xffffu) | ((tensor_d1 & 0xffffu) << 16));
    g1[3] = (int)(((tensor_d1 >> 16) & 0xffffu) | ((tile_d0 & 0xffffu) << 16));
    g1[4] = (int)(tile_d1 & 0xffffu);                      // tile_dim1 | tile_dim2=0
    g1[5] = (int)(unsigned)(stride0_elems & 0xffffffffu);  // dim0 stride lo
    g1[6] = (int)(unsigned)((stride0_elems >> 32) & 0xffffu); // stride hi | stride1=0
    g1[7] = 0;
    v4i gz4 = {0, 0, 0, 0};
    v8i gz8 = {0, 0, 0, 0, 0, 0, 0, 0};
    __builtin_amdgcn_tensor_load_to_lds(g0, g1, gz4, gz4, gz8, 0);
}

// ---------------------------------------------------------------------------
// Embedding gather + f32 -> bf16
// ---------------------------------------------------------------------------
__global__ void k_embed(const int* __restrict__ x, const float* __restrict__ emb,
                        bf16_t* __restrict__ act) {
    int row = blockIdx.x;            // t*B + b
    int t = row >> 5, b = row & 31;
    int tok = x[b * SS + t];
    const float* src = emb + (size_t)tok * HH;
    bf16_t* dst = act + (size_t)row * HH;
    for (int i = threadIdx.x; i < HH; i += blockDim.x)
        dst[i] = (bf16_t)src[i];
}

__global__ void k_zero_state(bf16_t* __restrict__ h, float* __restrict__ c) {
    int i = blockIdx.x * blockDim.x + threadIdx.x;
    if (i < BB * HH) { h[i] = (bf16_t)0.0f; c[i] = 0.0f; }
}

__global__ void k_loss_init(float* __restrict__ acc) { *acc = 0.0f; }

// ---------------------------------------------------------------------------
// Shared helper: stage a 64-row x 64-k chunk of f32 weights into bf16 LDS.
// ---------------------------------------------------------------------------
__device__ __forceinline__ void stage_w(bf16_t (*Ws)[KC + 8], const float* __restrict__ Wrow,
                                        int tid, int ks) {
    int r = tid >> 2, kp = (tid & 3) * 16;
    const float4* src = (const float4*)(Wrow + ks + kp);
    v8bf o0, o1; float4 f;
    f = src[0]; o0[0]=(bf16_t)f.x; o0[1]=(bf16_t)f.y; o0[2]=(bf16_t)f.z; o0[3]=(bf16_t)f.w;
    f = src[1]; o0[4]=(bf16_t)f.x; o0[5]=(bf16_t)f.y; o0[6]=(bf16_t)f.z; o0[7]=(bf16_t)f.w;
    f = src[2]; o1[0]=(bf16_t)f.x; o1[1]=(bf16_t)f.y; o1[2]=(bf16_t)f.z; o1[3]=(bf16_t)f.w;
    f = src[3]; o1[4]=(bf16_t)f.x; o1[5]=(bf16_t)f.y; o1[6]=(bf16_t)f.z; o1[7]=(bf16_t)f.w;
    *(v8bf*)&Ws[r][kp]     = o0;
    *(v8bf*)&Ws[r][kp + 8] = o1;
    if (ks + KC < HH) __builtin_prefetch((const void*)(Wrow + ks + KC + kp), 0, 1);
}

// ---------------------------------------------------------------------------
// Big input-projection GEMM:  X[4096][4096] = A_bf16[4096][1024] @ W[N][K]^T
//   + (bias0 + bias1).  A tile DMA'd by TDM; W converted f32->bf16 by VALU
//   concurrently with the TDM transfer.
// ---------------------------------------------------------------------------
__global__ void k_bigmm(const bf16_t* __restrict__ A, const float* __restrict__ W,
                        const float* __restrict__ bias0, const float* __restrict__ bias1,
                        float* __restrict__ C) {
    __shared__ bf16_t As[64][KC + 8];
    __shared__ bf16_t Ws[64][KC + 8];
    const int m0 = blockIdx.y * 64, n0 = blockIdx.x * 64;
    const int tid = threadIdx.x, wave = tid >> 5, lane = tid & 31;
    const int mt = wave >> 2;                 // 0..1  (m tiles mt, mt+2)
    const int nt = wave & 3;                  // 0..3
    const unsigned lds_as = (unsigned)(uintptr_t)(void*)&As[0][0];
    v8f acc0 = {}; v8f acc1 = {};

    for (int ks = 0; ks < HH; ks += KC) {
        if (wave == 0)   // async DMA of the bf16 A tile: overlaps W conversion
            tdm_load_tile_bf16(lds_as, A + (size_t)m0 * HH + ks,
                               HH, MR, KC, 64, HH);
        stage_w(Ws, W + (size_t)(n0 + (tid >> 2)) * HH, tid, ks);
        __builtin_amdgcn_s_wait_tensorcnt(0);
        __syncthreads();
        #pragma unroll
        for (int kk = 0; kk < KC; kk += 32) {
            ABFrag a0, a1, bfr;
            int ar0 = mt * 16 + (lane & 15);
            int ar1 = (mt + 2) * 16 + (lane & 15);
            int ab  = kk + ((lane >> 4) << 3);            // +0 / +8
            a0.h[0] = *(const v8bf*)&As[ar0][ab];
            a0.h[1] = *(const v8bf*)&As[ar0][ab + 16];
            a1.h[0] = *(const v8bf*)&As[ar1][ab];
            a1.h[1] = *(const v8bf*)&As[ar1][ab + 16];
            int br = nt * 16 + (lane & 15);
            int bb = kk + ((lane >> 4) << 3);             // +0 / +8
            bfr.h[0] = *(const v8bf*)&Ws[br][bb];
            bfr.h[1] = *(const v8bf*)&Ws[br][bb + 16];
            acc0 = __builtin_amdgcn_wmma_f32_16x16x32_bf16(false, a0.v, false, bfr.v,
                                                           (short)0, acc0, false, false);
            acc1 = __builtin_amdgcn_wmma_f32_16x16x32_bf16(false, a1.v, false, bfr.v,
                                                           (short)0, acc1, false, false);
        }
        __syncthreads();
    }
    int ncol = n0 + nt * 16 + (lane & 15);
    float bs = bias0[ncol] + bias1[ncol];
    int mb0 = m0 + mt * 16 + ((lane >> 4) << 3);
    int mb1 = m0 + (mt + 2) * 16 + ((lane >> 4) << 3);
    #pragma unroll
    for (int r = 0; r < 8; ++r) {
        C[(size_t)(mb0 + r) * G4 + ncol] = acc0[r] + bs;
        C[(size_t)(mb1 + r) * G4 + ncol] = acc1[r] + bs;
    }
}

// ---------------------------------------------------------------------------
// Decoder GEMM, remapped so M = timestep with batch fixed per block:
// lane fragment elements become 8 *consecutive t* -> float4 stores into the
// permuted [B, V, T] output (contiguous along T).
// grid = (V/64, T/64, B)
// ---------------------------------------------------------------------------
__global__ void k_decoder(const bf16_t* __restrict__ A, const float* __restrict__ W,
                          const float* __restrict__ bd, float* __restrict__ out) {
    __shared__ bf16_t As[64][KC + 8];
    __shared__ bf16_t Ws[64][KC + 8];
    const int n0 = blockIdx.x * 64;
    const int t0 = blockIdx.y * 64;
    const int bz = blockIdx.z;
    const int tid = threadIdx.x, wave = tid >> 5, lane = tid & 31;
    const int mt = wave >> 2, nt = wave & 3;
    const unsigned lds_as = (unsigned)(uintptr_t)(void*)&As[0][0];
    v8f acc0 = {}; v8f acc1 = {};

    // A tile rows: global row (t0+r)*32 + bz  ->  row stride 32*HH elements
    const bf16_t* Abase = A + ((size_t)t0 * BB + bz) * HH;

    for (int ks = 0; ks < HH; ks += KC) {
        if (wave == 0)
            tdm_load_tile_bf16(lds_as, Abase + ks,
                               HH, MR, KC, 64, (unsigned long long)BB * HH);
        stage_w(Ws, W + (size_t)(n0 + (tid >> 2)) * HH, tid, ks);
        __builtin_amdgcn_s_wait_tensorcnt(0);
        __syncthreads();
        #pragma unroll
        for (int kk = 0; kk < KC; kk += 32) {
            ABFrag a0, a1, bfr;
            int ar0 = mt * 16 + (lane & 15);
            int ar1 = (mt + 2) * 16 + (lane & 15);
            int ab  = kk + ((lane >> 4) << 3);
            a0.h[0] = *(const v8bf*)&As[ar0][ab];
            a0.h[1] = *(const v8bf*)&As[ar0][ab + 16];
            a1.h[0] = *(const v8bf*)&As[ar1][ab];
            a1.h[1] = *(const v8bf*)&As[ar1][ab + 16];
            int br = nt * 16 + (lane & 15);
            int bb = kk + ((lane >> 4) << 3);
            bfr.h[0] = *(const v8bf*)&Ws[br][bb];
            bfr.h[1] = *(const v8bf*)&Ws[br][bb + 16];
            acc0 = __builtin_amdgcn_wmma_f32_16x16x32_bf16(false, a0.v, false, bfr.v,
                                                           (short)0, acc0, false, false);
            acc1 = __builtin_amdgcn_wmma_f32_16x16x32_bf16(false, a1.v, false, bfr.v,
                                                           (short)0, acc1, false, false);
        }
        __syncthreads();
    }
    int v = n0 + nt * 16 + (lane & 15);
    float bs = bd[v];
    int tb0 = t0 + mt * 16 + ((lane >> 4) << 3);
    int tb1 = t0 + (mt + 2) * 16 + ((lane >> 4) << 3);
    float* p0 = out + (size_t)bz * VV * TT + (size_t)v * TT + tb0;
    float* p1 = out + (size_t)bz * VV * TT + (size_t)v * TT + tb1;
    float4 q;
    q.x = acc0[0] + bs; q.y = acc0[1] + bs; q.z = acc0[2] + bs; q.w = acc0[3] + bs;
    *(float4*)p0 = q;
    q.x = acc0[4] + bs; q.y = acc0[5] + bs; q.z = acc0[6] + bs; q.w = acc0[7] + bs;
    *(float4*)(p0 + 4) = q;
    q.x = acc1[0] + bs; q.y = acc1[1] + bs; q.z = acc1[2] + bs; q.w = acc1[3] + bs;
    *(float4*)p1 = q;
    q.x = acc1[4] + bs; q.y = acc1[5] + bs; q.z = acc1[6] + bs; q.w = acc1[7] + bs;
    *(float4*)(p1 + 4) = q;
}

// ---------------------------------------------------------------------------
// Fused recurrent step: gates = X[t] + h @ Whh^T (WMMA), LSTM cell (fp32),
// BatchNorm over batch + locked dropout. Block = 16 channels x 4 gates x B=32.
// ---------------------------------------------------------------------------
__global__ void k_step(const bf16_t* __restrict__ hin, bf16_t* __restrict__ hout,
                       float* __restrict__ cstate,
                       const float* __restrict__ X, const float* __restrict__ Whh,
                       const float* __restrict__ gamma, const float* __restrict__ beta,
                       const float* __restrict__ mask,
                       bf16_t* __restrict__ act, int t) {
    __shared__ bf16_t As[32][KC + 8];
    __shared__ bf16_t Ws[64][KC + 8];
    __shared__ float  gl[4][32][16];
    __shared__ float  hl[32][17];
    __shared__ float  mu_s[16], rv_s[16];

    const int c0 = blockIdx.x * 16;
    const int tid = threadIdx.x, wave = tid >> 5, lane = tid & 31;
    const int gw = wave >> 1;        // gate 0..3
    const int mh = wave & 1;         // batch half
    v8f acc = {};

    for (int ks = 0; ks < HH; ks += KC) {
        { // stage h (bf16): 32 rows x 64 k
            int r = tid >> 3, kp = (tid & 7) * 8;
            const uint4* src = (const uint4*)(hin + (size_t)r * HH + ks + kp);
            *(uint4*)&As[r][kp] = src[0];
        }
        { // stage Whh rows for the 4 gate slices, f32 -> bf16
            int rr = tid >> 2;
            int grow = (rr >> 4) * HH + c0 + (rr & 15);
            stage_w(Ws, Whh + (size_t)grow * HH, tid, ks);
        }
        __syncthreads();
        #pragma unroll
        for (int kk = 0; kk < KC; kk += 32) {
            ABFrag a, bfr;
            int ar = mh * 16 + (lane & 15);
            int ab = kk + ((lane >> 4) << 3);
            a.h[0] = *(const v8bf*)&As[ar][ab];
            a.h[1] = *(const v8bf*)&As[ar][ab + 16];
            int br = gw * 16 + (lane & 15);
            int bb = kk + ((lane >> 4) << 3);
            bfr.h[0] = *(const v8bf*)&Ws[br][bb];
            bfr.h[1] = *(const v8bf*)&Ws[br][bb + 16];
            acc = __builtin_amdgcn_wmma_f32_16x16x32_bf16(false, a.v, false, bfr.v,
                                                          (short)0, acc, false, false);
        }
        __syncthreads();
    }

    { // add precomputed input projection X[t], park gates in LDS
        int ch = lane & 15;
        int gcol = gw * HH + c0 + ch;
        int bbase = mh * 16 + ((lane >> 4) << 3);
        const float* Xrow = X + (size_t)(t * BB) * G4;
        #pragma unroll
        for (int r = 0; r < 8; ++r) {
            int b = bbase + r;
            gl[gw][b][ch] = acc[r] + Xrow[(size_t)b * G4 + gcol];
        }
    }
    __syncthreads();

    // LSTM cell (fp32 state)
    for (int idx = tid; idx < 512; idx += 256) {
        int b = idx & 31, ch = idx >> 5;
        float ig = sigf(gl[0][b][ch]);
        float fg = sigf(gl[1][b][ch]);
        float gg = tanhf(gl[2][b][ch]);
        float og = sigf(gl[3][b][ch]);
        float* cp = cstate + (size_t)b * HH + c0 + ch;
        float cn = fg * (*cp) + ig * gg;
        *cp = cn;
        float h = og * tanhf(cn);
        hl[b][ch] = h;
        hout[(size_t)b * HH + c0 + ch] = (bf16_t)h;
    }
    __syncthreads();

    // BatchNorm stats over batch (B=32 fully inside block)
    if (tid < 16) {
        float s = 0.f, s2 = 0.f;
        for (int b = 0; b < 32; ++b) { float v = hl[b][tid]; s += v; s2 += v * v; }
        float m = s * (1.0f / 32.0f);
        float var = s2 * (1.0f / 32.0f) - m * m;
        mu_s[tid] = m;
        rv_s[tid] = rsqrtf(var + 1e-5f);
    }
    __syncthreads();

    // BN affine + locked dropout (scale 1/(1-0.3)), write bf16 seq output
    for (int idx = tid; idx < 512; idx += 256) {
        int b = idx & 31, ch = idx >> 5;
        float h = hl[b][ch];
        float a = (gamma[c0 + ch] * (h - mu_s[ch]) * rv_s[ch] + beta[c0 + ch])
                  * mask[(size_t)b * HH + c0 + ch] * (1.0f / 0.7f);
        act[((size_t)(t * BB + b)) * HH + c0 + ch] = (bf16_t)a;
    }
}

// ---------------------------------------------------------------------------
// Online-softmax NLL over permuted logits [B, V, T]; threadIdx.x = t keeps
// reads coalesced along the contiguous T axis.
// ---------------------------------------------------------------------------
__global__ void k_loss(const float* __restrict__ logits, const int* __restrict__ x,
                       float* __restrict__ acc) {
    __shared__ float red[128];
    int t = threadIdx.x;       // 0..127
    int b = blockIdx.x;        // 0..31
    const float* base = logits + (size_t)b * VV * TT + t;
    float mx = -INFINITY, sm = 0.0f;
    for (int v = 0; v < VV; ++v) {
        float z = base[(size_t)v * TT];
        if (z > mx) { sm = sm * __expf(mx - z) + 1.0f; mx = z; }
        else        { sm += __expf(z - mx); }
    }
    int tgt = x[b * SS + t + 1];
    float zt = base[(size_t)tgt * TT];
    float nll = (mx + __logf(sm)) - zt;
    red[t] = nll;
    __syncthreads();
    for (int s = 64; s > 0; s >>= 1) {
        if (t < s) red[t] += red[t + s];
        __syncthreads();
    }
    if (t == 0) atomicAdd(acc, red[0]);
}

__global__ void k_loss_fin(const float* __restrict__ acc, float* __restrict__ out) {
    out[0] = acc[0] / (float)(TT * BB);
}

// ---------------------------------------------------------------------------
extern "C" void kernel_launch(void* const* d_in, const int* in_sizes, int n_in,
                              void* d_out, int out_size, void* d_ws, size_t ws_size,
                              hipStream_t stream) {
    (void)in_sizes; (void)n_in; (void)out_size; (void)ws_size;

    const int*   x   = (const int*)d_in[0];
    const float* emb = (const float*)d_in[1];
    const float* Wd  = (const float*)d_in[2];
    const float* bd  = (const float*)d_in[3];
    const float* Wih[3]  = {(const float*)d_in[4],  (const float*)d_in[11], (const float*)d_in[18]};
    const float* Whh[3]  = {(const float*)d_in[5],  (const float*)d_in[12], (const float*)d_in[19]};
    const float* bih[3]  = {(const float*)d_in[6],  (const float*)d_in[13], (const float*)d_in[20]};
    const float* bhh[3]  = {(const float*)d_in[7],  (const float*)d_in[14], (const float*)d_in[21]};
    const float* gam[3]  = {(const float*)d_in[8],  (const float*)d_in[15], (const float*)d_in[22]};
    const float* bet[3]  = {(const float*)d_in[9],  (const float*)d_in[16], (const float*)d_in[23]};
    const float* msk[3]  = {(const float*)d_in[10], (const float*)d_in[17], (const float*)d_in[24]};

    float* out = (float*)d_out;
    float* logits = out + 1;   // [B, V, T]

    // workspace carve-up (~72.25 MB)
    char* w = (char*)d_ws;
    float*  X    = (float*)w;   w += (size_t)MR * G4 * sizeof(float);     // 64 MB
    bf16_t* act  = (bf16_t*)w;  w += (size_t)MR * HH * sizeof(bf16_t);    // 8 MB
    bf16_t* h0   = (bf16_t*)w;  w += (size_t)BB * HH * sizeof(bf16_t);
    bf16_t* h1   = (bf16_t*)w;  w += (size_t)BB * HH * sizeof(bf16_t);
    float*  cst  = (float*)w;   w += (size_t)BB * HH * sizeof(float);
    float*  lacc = (float*)w;   w += sizeof(float);

    // 1) embedding gather -> bf16 activations
    k_embed<<<MR, 256, 0, stream>>>(x, emb, act);

    // 2) three LSTM layers: batched input projection, then fused recurrence
    for (int l = 0; l < 3; ++l) {
        k_bigmm<<<dim3(G4 / 64, MR / 64), 256, 0, stream>>>(act, Wih[l], bih[l], bhh[l], X);
        k_zero_state<<<(BB * HH + 255) / 256, 256, 0, stream>>>(h0, cst);
        for (int t = 0; t < TT; ++t) {
            bf16_t* hin  = (t & 1) ? h1 : h0;
            bf16_t* hout = (t & 1) ? h0 : h1;
            k_step<<<HH / 16, 256, 0, stream>>>(hin, hout, cst, X, Whh[l],
                                                gam[l], bet[l], msk[l], act, t);
        }
    }

    // 3) decoder GEMM straight into permuted [B, V, T] output
    k_decoder<<<dim3(VV / 64, TT / 64, BB), 256, 0, stream>>>(act, Wd, bd, logits);

    // 4) NLL loss
    k_loss_init<<<1, 1, 0, stream>>>(lacc);
    k_loss<<<BB, TT, 0, stream>>>(logits, x, lacc);
    k_loss_fin<<<1, 1, 0, stream>>>(lacc, out);
}
